// my_Conv2d_69123203662203
// MI455X (gfx1250) — matrix operand order repaired and checked by reading, not compiled, not run on hardware
//
#include <hip/hip_runtime.h>

// Quantized 3x3 conv (pad=1, stride=1) as implicit GEMM on CDNA5 WMMA.
//   M = Cout = 128, N = B*H*W = 200704, K = Cin*3*3 = 576
// K is reordered tap-major: k' = (kh*3+kw)*64 + ci, so each K-chunk of 64
// has one fixed filter tap -> im2col staging is a constant-stride channel
// loop with a single bounds check (no per-element div/mod). Activation
// loads for tap t+1 are prefetched into registers during tap t's WMMA
// phase, hiding L2 latency behind the matrix pipe.
// Weights are pre-quantized to f16 in d_ws (tap-major) by a pre-pass; the
// A-fragments are then direct global b128 loads (L2-resident, 144 KB).
// xq=trunc(32x), wq=trunc(256w) are small ints -> exact in f16/f32-acc WMMA.

typedef __attribute__((ext_vector_type(16))) _Float16 v16h;
typedef __attribute__((ext_vector_type(8)))  _Float16 v8h;
typedef __attribute__((ext_vector_type(8)))  float    v8f;

#define CIN    64
#define COUT   128
#define HW     112
#define KDIM   576          // 64 * 9
#define NTILE  128          // output pixels per block
#define SAPAD  72           // 64 + 8 halves pad -> conflict-free b128 reads
#define HW2    (HW * HW)    // 12544

union Frag16 { v16h v; v8h h[2]; };

// ---- pre-pass: quantize + reorder weights to tap-major f16 ----------------
// wq[m*576 + tap*64 + ci] = trunc(256 * w[m*576 + ci*9 + tap])
__global__ __launch_bounds__(64)
void quant_weights(const float* __restrict__ w, _Float16* __restrict__ wq)
{
    const int m   = blockIdx.x;     // 0..127
    const int tap = blockIdx.y;     // 0..8
    const int ci  = threadIdx.x;    // 0..63
    const float v = w[m * KDIM + ci * 9 + tap];
    wq[m * KDIM + tap * CIN + ci] = (_Float16)truncf(v * 256.0f);
}

// ---- main kernel -----------------------------------------------------------
template <bool PREQ>
__global__ __launch_bounds__(256)
void qconv3x3_wmma_f16(const float* __restrict__ x,
                       const _Float16* __restrict__ wq,   // PREQ path (d_ws)
                       const float* __restrict__ wraw,    // fallback path
                       const float* __restrict__ bias,
                       float* __restrict__ out)
{
    __shared__ _Float16 sA[NTILE * SAPAD];   // act tile [n][ci], 18 KB

    const int t    = threadIdx.x;
    const int lane = t & 31;
    const int wave = t >> 5;                 // 0..7 -> M-chunk of 16

    // staging mapping: 128 consecutive threads = 128 pixels (coalesced in ow)
    const int nloc  = t & 127;
    const int chalf = t >> 7;                // 0/1 -> ci 0..31 / 32..63
    const int ng    = blockIdx.x * NTILE + nloc;
    const int ow    = ng % HW;
    const int oh    = (ng / HW) % HW;
    const int bb    = ng / HW2;

    // epilogue rows (C/D layout: M = mbase + r)
    const int mbase = wave * 16 + (lane >> 4) * 8;
    float bv[8];
    #pragma unroll
    for (int r = 0; r < 8; ++r) bv[r] = bias[mbase + r];

    // A-fragment addressing (16-bit 16x32 layout):
    // lane<16: K ko+0..7 / ko+16..23 with ko=0 ; lane>=16: ko=8
    const int am  = wave * 16 + (lane & 15);
    const int ako = (lane >> 4) * 8;

    // prefetch one tap's worth of activations (32 strided channel loads)
    float pf[32];
    auto load_tap = [&](int tap) {
        const int kh = tap / 3;              // uniform SALU math
        const int kv = tap - kh * 3;
        const int ih = oh - 1 + kh;
        const int iw = ow - 1 + kv;
        const bool inb = ((unsigned)ih < (unsigned)HW) &&
                         ((unsigned)iw < (unsigned)HW);
        const float* src =
            x + (((size_t)bb * CIN + chalf * 32) * HW + ih) * HW + iw;
        #pragma unroll
        for (int j = 0; j < 32; ++j)
            pf[j] = inb ? src[(size_t)j * HW2] : 0.0f;   // immediate offsets
    };

    v8f acc[8] = {};
    load_tap(0);

    for (int tap = 0; tap < 9; ++tap) {
        __syncthreads();   // previous chunk's fragment reads done

        // quantize prefetched channels: trunc(32*x) -> f16, 4x b128 stores
        {
            _Float16* dst = &sA[nloc * SAPAD + chalf * 32];
            #pragma unroll
            for (int g = 0; g < 4; ++g) {
                union { _Float16 h[8]; v8h v; } pk;
                #pragma unroll
                for (int i = 0; i < 8; ++i)
                    pk.h[i] = (_Float16)truncf(pf[g * 8 + i] * 32.0f);
                *(v8h*)&dst[g * 8] = pk.v;
            }
        }

        __syncthreads();   // tile visible

        // issue next tap's global loads now; they complete under the WMMAs
        if (tap < 8) load_tap(tap + 1);

        // two K=32 sub-steps over the 64 staged channels
        #pragma unroll
        for (int ks = 0; ks < 64; ks += 32) {
            Frag16 a;
            if constexpr (PREQ) {
                const _Float16* ap = wq + am * KDIM + tap * 64 + ks + ako;
                a.h[0] = *(const v8h*)(ap);        // K ako+0..7
                a.h[1] = *(const v8h*)(ap + 16);   // K ako+16..23
            } else {
                const float* wp = wraw + am * KDIM + tap;  // stride-9 gather
                #pragma unroll
                for (int i = 0; i < 8; ++i) {
                    a.h[0][i] = (_Float16)truncf(wp[(ks + ako + i)      * 9] * 256.0f);
                    a.h[1][i] = (_Float16)truncf(wp[(ks + ako + 16 + i) * 9] * 256.0f);
                }
            }
            const int kb = ks + (lane >> 4) * 8;   // B: K kb+0..7 / kb+16..23 (same as A)
            #pragma unroll
            for (int nt = 0; nt < 8; ++nt) {
                Frag16 b;
                const int n = nt * 16 + (lane & 15);
                b.h[0] = *(v8h*)&sA[n * SAPAD + kb];
                b.h[1] = *(v8h*)&sA[n * SAPAD + kb + 16];
                acc[nt] = __builtin_amdgcn_wmma_f32_16x16x32_f16(
                    false, a.v, false, b.v, (short)0, acc[nt], false, false);
            }
        }
    }

    // ---- epilogue: dequant (2^-13), add bias, scatter to NCHW --------------
    const float scale = 1.0f / 8192.0f;
    #pragma unroll
    for (int nt = 0; nt < 8; ++nt) {
        const int n   = blockIdx.x * NTILE + nt * 16 + (lane & 15);
        const int oww = n % HW;
        const int ohh = (n / HW) % HW;
        const int bbb = n / HW2;
        float* orow = out + (((size_t)bbb * COUT + mbase) * HW + ohh) * HW + oww;
        #pragma unroll
        for (int r = 0; r < 8; ++r)
            orow[(size_t)r * HW2] = acc[nt][r] * scale + bv[r];
    }
}

extern "C" void kernel_launch(void* const* d_in, const int* in_sizes, int n_in,
                              void* d_out, int out_size, void* d_ws, size_t ws_size,
                              hipStream_t stream) {
    const float* x    = (const float*)d_in[0];   // (16, 64, 112, 112)
    const float* wgt  = (const float*)d_in[1];   // (128, 64, 3, 3)
    const float* bias = (const float*)d_in[2];   // (128,)
    float* out        = (float*)d_out;           // (16, 128, 112, 112)

    const int npix = 16 * HW2;                   // 200704, divisible by 128
    dim3 grid(npix / NTILE);                     // 1568 blocks

    const size_t wq_bytes = (size_t)COUT * KDIM * sizeof(_Float16);  // 144 KB
    if (ws_size >= wq_bytes) {
        _Float16* wq = (_Float16*)d_ws;
        quant_weights<<<dim3(COUT, 9), 64, 0, stream>>>(wgt, wq);
        qconv3x3_wmma_f16<true><<<grid, 256, 0, stream>>>(x, wq, wgt, bias, out);
    } else {
        qconv3x3_wmma_f16<false><<<grid, 256, 0, stream>>>(x, nullptr, wgt, bias, out);
    }
}
